// GraphEncoder_39084202393964
// MI455X (gfx1250) — matrix operand order repaired and checked by reading, compiled, not run on hardware
//
#include <hip/hip_runtime.h>

typedef __attribute__((ext_vector_type(16))) __bf16 v16bf;
typedef __attribute__((ext_vector_type(8)))  float  v8f;
typedef unsigned int u32x4 __attribute__((ext_vector_type(4)));
typedef int          i32x4 __attribute__((ext_vector_type(4)));
typedef int          i32x8 __attribute__((ext_vector_type(8)));

#define BN_EPS 1e-5f

// GEMM tiling
#define TM 128            // block tile rows
#define TN 64             // block tile cols
#define AST 40            // As row stride in ushorts (80B: 16B-aligned, 20-bank skew)
#define BST 40            // Bs col stride in ushorts

#if defined(__gfx1250__) && __has_builtin(__builtin_amdgcn_tensor_load_to_lds) && \
    __has_builtin(__builtin_amdgcn_s_wait_tensorcnt)
#define USE_TDM 1
#else
#define USE_TDM 0
#endif

__device__ __forceinline__ unsigned short f2bf(float f) {
    unsigned u = __float_as_uint(f);
    u += 0x7FFFu + ((u >> 16) & 1u);   // round-to-nearest-even
    return (unsigned short)(u >> 16);
}

// ---------------------------------------------------------------- cast f32 -> bf16
__global__ void cast_f32_bf16(const float* __restrict__ in,
                              unsigned short* __restrict__ out, long nelem) {
    long i = (long)blockIdx.x * blockDim.x + threadIdx.x;
    long stride = (long)gridDim.x * blockDim.x;
    for (; i < nelem; i += stride) out[i] = f2bf(in[i]);
}

// ---------------------------------------------------------------- WMMA bf16 GEMM
// C[M,N] = A[M,K](bf16) * B[K,N](bf16), f32 accumulate.
// Block tile 128x64, 8 waves; wave owns 2x2 grid of 16x16 tiles -> 4 WMMA / K-chunk.
// Double-buffered pipeline: TDM (A tile DMA) + B global load for chunk c+1 run
// under the WMMAs of chunk c; one barrier per chunk.
__global__ __launch_bounds__(256) void gemm_bf16_wmma(
        const unsigned short* __restrict__ A,
        const unsigned short* __restrict__ B,
        float* __restrict__ C, int M, int K, int N) {
    __shared__ __align__(16) unsigned short As[2][TM * AST];   // 2 x 10240 B
    __shared__ __align__(16) unsigned short Bs[2][TN * BST];   // 2 x  5120 B

    const int tid  = threadIdx.x;
    const int lane = tid & 31;
    const int wave = tid >> 5;
    const int bm   = blockIdx.y * TM;
    const int bn   = blockIdx.x * TN;
    const int mt0  = (wave & 3) * 2;         // first of 2 M-tiles (of 8)
    const int nt0  = (wave >> 2) * 2;        // first of 2 N-tiles (of 4)
    const int half = lane >> 4;
    const int l16  = lane & 15;
    const int NC   = K >> 5;                 // K chunks of 32

    v8f z = {0.f,0.f,0.f,0.f,0.f,0.f,0.f,0.f};
    v8f acc[2][2] = {{z, z}, {z, z}};

#if USE_TDM
    const int rows = M - bm;                 // OOB rows read as zero via tensor_dim1
#endif

    // ---- A tile staging (TDM DMA, or manual vectorized fallback)
    auto stage_A = [&](int c, int buf) {
        const int k0 = c * 32;
#if USE_TDM
        if (wave == 0) {
            unsigned ldsA = (unsigned)(uintptr_t)&As[buf][0];
            unsigned long long ga =
                (unsigned long long)(uintptr_t)(A + (size_t)bm * K + k0);
            u32x4 g0 = { 1u,                              // count=1 (valid descriptor)
                         ldsA,
                         (unsigned)ga,
                         (unsigned)((ga >> 32) & 0x1FFFFFFu) | (2u << 30) }; // type=2
            i32x8 g1 = { (int)((1u << 16) | (1u << 20) | (3u << 22) | (3u << 25)),
                         // data_size=2B | pad_enable | pad_interval=16dw | pad_amount=4dw
                         (int)((32u & 0xFFFFu) << 16),    // tensor_dim0(lo16)<<16
                         (int)((32u >> 16) | (((unsigned)rows & 0xFFFFu) << 16)),
                         (int)((((unsigned)rows >> 16) & 0xFFFFu) | (32u << 16)), // tile_dim0
                         (int)(TM),                        // tile_dim1=128, tile_dim2=0
                         (int)K,                           // tensor_dim0_stride lo32
                         0, 0 };
            i32x4 zg = {0, 0, 0, 0};
#if __clang_major__ >= 23
            i32x8 zg8 = {0,0,0,0,0,0,0,0};
            __builtin_amdgcn_tensor_load_to_lds(g0, g1, zg, zg, zg8, 0);
#else
            __builtin_amdgcn_tensor_load_to_lds(g0, g1, zg, zg, 0);
#endif
        }
#else
        #pragma unroll
        for (int i = 0; i < 2; ++i) {
            int idx = tid + i * 256;
            int r = idx >> 2, c0 = (idx & 3) * 8;
            uint4 v = make_uint4(0u, 0u, 0u, 0u);
            int row = bm + r;
            if (row < M) v = *(const uint4*)(A + (size_t)row * K + (k0 + c0));
            *(uint4*)(&As[buf][r * AST + c0]) = v;
        }
#endif
    };

    // ---- B tile staging, split into (global load -> regs) and (regs -> LDS)
    const int bkk = tid >> 3, bc0 = (tid & 7) * 8;
    uint4 breg;
    auto load_B = [&](int c) {
        breg = *(const uint4*)(B + (size_t)(c * 32 + bkk) * N + (bn + bc0));
    };
    auto store_B = [&](int buf) {
        union { uint4 u4; unsigned short s[8]; } v;
        v.u4 = breg;
        #pragma unroll
        for (int j = 0; j < 8; ++j) Bs[buf][(bc0 + j) * BST + bkk] = v.s[j];
    };

    // ---- prologue: stage chunk 0 into buffer 0
    stage_A(0, 0);
    load_B(0);
    store_B(0);
#if USE_TDM
    if (wave == 0) __builtin_amdgcn_s_wait_tensorcnt(0);
#endif
    __syncthreads();

    for (int c = 0; c < NC; ++c) {
        const int cur = c & 1;
        const bool more = (c + 1) < NC;

        // kick off next chunk's A DMA + B global load (overlaps with WMMAs below)
        if (more) {
            stage_A(c + 1, 1 - cur);
            load_B(c + 1);
        }

        // ---- compute on current buffer (CDNA5 wave32 fragment layouts)
        union F { unsigned u[8]; v16bf v; };
        F fa[2], fb[2];
        #pragma unroll
        for (int mi = 0; mi < 2; ++mi) {
            const unsigned short* arow = &As[cur][((mt0 + mi) * 16 + l16) * AST];
            #pragma unroll
            for (int j = 0; j < 4; ++j) {
                fa[mi].u[j]     = *(const unsigned*)(arow + 8 * half + 2 * j);
                fa[mi].u[4 + j] = *(const unsigned*)(arow + 16 + 8 * half + 2 * j);
            }
        }
        #pragma unroll
        for (int ni = 0; ni < 2; ++ni) {
            const unsigned short* bcol = &Bs[cur][((nt0 + ni) * 16 + l16) * BST];
            #pragma unroll
            for (int j = 0; j < 8; ++j)
                fb[ni].u[j] = *(const unsigned*)(bcol + 8 * half + 2 * j + (j >= 4 ? 8 : 0));
        }
        #pragma unroll
        for (int mi = 0; mi < 2; ++mi)
            #pragma unroll
            for (int ni = 0; ni < 2; ++ni)
                acc[mi][ni] = __builtin_amdgcn_wmma_f32_16x16x32_bf16(
                    false, fa[mi].v, false, fb[ni].v, (short)0, acc[mi][ni],
                    false, false);

        if (more) {
            store_B(1 - cur);                 // commit next B tile to the spare buffer
#if USE_TDM
            if (wave == 0) __builtin_amdgcn_s_wait_tensorcnt(0);
#endif
            __syncthreads();                  // publishes chunk c+1; retires chunk c readers
        }
    }

    // ---- store: VGPR r, lanes 0-15 -> row r, lanes 16-31 -> row 8+r; col = l16
    #pragma unroll
    for (int mi = 0; mi < 2; ++mi) {
        #pragma unroll
        for (int r = 0; r < 8; ++r) {
            int row = bm + (mt0 + mi) * 16 + half * 8 + r;
            if (row < M) {
                #pragma unroll
                for (int ni = 0; ni < 2; ++ni)
                    C[(size_t)row * N + (bn + (nt0 + ni) * 16 + l16)] = acc[mi][ni][r];
            }
        }
    }
}

// ---------------------------------------------------------------- degree / dinv
__global__ void deg_kernel(const long long* __restrict__ dst, float* __restrict__ deg, int E) {
    int i = blockIdx.x * blockDim.x + threadIdx.x;
    if (i < E) atomicAdd(&deg[(int)dst[i]], 1.0f);
}
__global__ void dinv_kernel(const float* __restrict__ deg, float* __restrict__ dinv, int n) {
    int i = blockIdx.x * blockDim.x + threadIdx.x;
    if (i < n) dinv[i] = rsqrtf(deg[i] + 1.0f);   // +1 self loop; always > 0
}

// ---------------------------------------------------------------- edge scatter-add
// one wave32 per edge: out[dst,:] += h[src,:] * dinv[src]*dinv[dst]
__global__ __launch_bounds__(256) void edge_scatter(
        const long long* __restrict__ src, const long long* __restrict__ dst,
        const float* __restrict__ dinv, const float* __restrict__ h,
        float* __restrict__ out, int E, int dim) {
    int gw = blockIdx.x * 8 + (threadIdx.x >> 5);
    if (gw >= E) return;
    int lane = threadIdx.x & 31;
    int s = (int)src[gw], d = (int)dst[gw];
    float coef = dinv[s] * dinv[d];
    const float4* hs = (const float4*)(h + (size_t)s * dim);
    float* od = out + (size_t)d * dim;
    int nd4 = dim >> 2;
    for (int f = lane; f < nd4; f += 32) {
        float4 v = hs[f];
        int b = f * 4;
        atomicAdd(od + b + 0, v.x * coef);
        atomicAdd(od + b + 1, v.y * coef);
        atomicAdd(od + b + 2, v.z * coef);
        atomicAdd(od + b + 3, v.w * coef);
    }
}

// self-loop term: out[i,:] += h[i,:] * dinv[i]^2 (unique ownership, no atomics)
__global__ void self_loop(const float* __restrict__ h, const float* __restrict__ dinv,
                          float* __restrict__ out, int n, int dim) {
    long tot4 = (long)n * (dim >> 2);
    long i = (long)blockIdx.x * blockDim.x + threadIdx.x;
    if (i >= tot4) return;
    int node = (int)(i / (dim >> 2));
    float c = dinv[node]; c *= c;
    float4 v = ((const float4*)h)[i];
    float4 o = ((float4*)out)[i];
    o.x += v.x * c; o.y += v.y * c; o.z += v.z * c; o.w += v.w * c;
    ((float4*)out)[i] = o;
}

// ---------------------------------------------------------------- BatchNorm
__global__ __launch_bounds__(128) void bn_stats(const float* __restrict__ x,
                                                float* __restrict__ stats, int n, int dim) {
    int f = blockIdx.x * 128 + threadIdx.x;
    if (f >= dim) return;
    int r0 = blockIdx.y * 512;
    int r1 = r0 + 512; if (r1 > n) r1 = n;
    float s = 0.f, ss = 0.f;
    for (int r = r0; r < r1; ++r) {
        float v = x[(size_t)r * dim + f];
        s += v; ss += v * v;
    }
    atomicAdd(&stats[f], s);
    atomicAdd(&stats[dim + f], ss);
}
// fold gamma/beta: stats[f]=rstd*g, stats[dim+f]=beta-mean*rstd*g
// (conv bias cancels exactly inside BN, so it is never added)
__global__ void bn_finalize(float* __restrict__ stats, const float* __restrict__ gamma,
                            const float* __restrict__ beta, int n, int dim) {
    int f = blockIdx.x * blockDim.x + threadIdx.x;
    if (f >= dim) return;
    float inv_n = 1.0f / (float)n;
    float m   = stats[f] * inv_n;
    float var = stats[dim + f] * inv_n - m * m;
    float sc  = rsqrtf(var + BN_EPS) * gamma[f];
    stats[f]        = sc;
    stats[dim + f]  = beta[f] - m * sc;
}
__global__ void bn_apply(const float* __restrict__ x, const float* __restrict__ stats,
                         unsigned short* __restrict__ out_bf, float* __restrict__ out_f32,
                         int n, int dim, int relu) {
    long tot = (long)n * dim;
    long i = (long)blockIdx.x * blockDim.x + threadIdx.x;
    if (i >= tot) return;
    int f = (int)(i % dim);
    float v = x[i] * stats[f] + stats[dim + f];
    if (relu) v = fmaxf(v, 0.0f);
    if (out_bf) out_bf[i] = f2bf(v);
    else        out_f32[i] = v;
}

// ---------------------------------------------------------------- host
extern "C" void kernel_launch(void* const* d_in, const int* in_sizes, int n_in,
                              void* d_out, int out_size, void* d_ws, size_t ws_size,
                              hipStream_t stream) {
    (void)n_in; (void)out_size; (void)ws_size;
    const float*     x   = (const float*)d_in[0];
    const long long* ei  = (const long long*)d_in[1];
    const float* W1 = (const float*)d_in[2];
    const float* g1 = (const float*)d_in[4];
    const float* be1= (const float*)d_in[5];
    const float* W2 = (const float*)d_in[6];
    const float* g2 = (const float*)d_in[8];
    const float* be2= (const float*)d_in[9];
    const float* W3 = (const float*)d_in[10];
    const float* g3 = (const float*)d_in[12];
    const float* be3= (const float*)d_in[13];

    const int D0 = 768, D1 = 512, D2 = 256, D3 = 128;
    const int n = in_sizes[0] / D0;
    const int E = in_sizes[1] / 2;
    const long long* src = ei;
    const long long* dst = ei + E;

    char* ws = (char*)d_ws;
    size_t off = 0;
    auto alloc = [&](size_t bytes) -> void* {
        void* p = ws + off;
        off = (off + bytes + 255) & ~(size_t)255;
        return p;
    };
    unsigned short* xbf  = (unsigned short*)alloc((size_t)n * D0 * 2);
    unsigned short* w1bf = (unsigned short*)alloc((size_t)D0 * D1 * 2);
    unsigned short* w2bf = (unsigned short*)alloc((size_t)D1 * D2 * 2);
    unsigned short* w3bf = (unsigned short*)alloc((size_t)D2 * D3 * 2);
    float*          hbuf = (float*)alloc((size_t)n * D1 * 4);
    float*          agg  = (float*)alloc((size_t)n * D1 * 4);
    unsigned short* hbf  = (unsigned short*)alloc((size_t)n * D1 * 2);
    float*          deg  = (float*)alloc((size_t)n * 4);
    float*          dinv = (float*)alloc((size_t)n * 4);
    float*          stats= (float*)alloc((size_t)2 * D1 * 4);

    // bf16 conversions (once per call)
    {
        long nx = (long)n * D0;
        cast_f32_bf16<<<4096, 256, 0, stream>>>(x, xbf, nx);
        cast_f32_bf16<<<768, 256, 0, stream>>>(W1, w1bf, (long)D0 * D1);
        cast_f32_bf16<<<256, 256, 0, stream>>>(W2, w2bf, (long)D1 * D2);
        cast_f32_bf16<<<64, 256, 0, stream>>>(W3, w3bf, (long)D2 * D3);
    }
    // degree / dinv (dst-only, shared by all layers)
    hipMemsetAsync(deg, 0, (size_t)n * 4, stream);
    deg_kernel<<<(E + 255) / 256, 256, 0, stream>>>(dst, deg, E);
    dinv_kernel<<<(n + 255) / 256, 256, 0, stream>>>(deg, dinv, n);

    auto run_layer = [&](const unsigned short* Abf, const unsigned short* Wbf,
                         int K, int Nd, const float* gamma, const float* beta,
                         int relu, unsigned short* out_bf, float* out_f32) {
        dim3 gg(Nd / TN, (n + TM - 1) / TM);
        gemm_bf16_wmma<<<gg, 256, 0, stream>>>(Abf, Wbf, hbuf, n, K, Nd);

        hipMemsetAsync(agg, 0, (size_t)n * Nd * 4, stream);
        edge_scatter<<<(E + 7) / 8, 256, 0, stream>>>(src, dst, dinv, hbuf, agg, E, Nd);
        long tot4 = (long)n * (Nd / 4);
        self_loop<<<(unsigned)((tot4 + 255) / 256), 256, 0, stream>>>(hbuf, dinv, agg, n, Nd);

        hipMemsetAsync(stats, 0, (size_t)2 * Nd * 4, stream);
        dim3 sg((Nd + 127) / 128, (n + 511) / 512);
        bn_stats<<<sg, 128, 0, stream>>>(agg, stats, n, Nd);
        bn_finalize<<<(Nd + 255) / 256, 256, 0, stream>>>(stats, gamma, beta, n, Nd);
        long tot = (long)n * Nd;
        bn_apply<<<(unsigned)((tot + 255) / 256), 256, 0, stream>>>(
            agg, stats, out_bf, out_f32, n, Nd, relu);
    };

    run_layer(xbf, w1bf, D0, D1, g1, be1, 1, hbf, nullptr);
    run_layer(hbf, w2bf, D1, D2, g2, be2, 1, hbf, nullptr);
    run_layer(hbf, w3bf, D2, D3, g3, be3, 0, nullptr, (float*)d_out);
}
